// HDModel_9105330668031
// MI455X (gfx1250) — matrix operand order repaired and checked by reading, not compiled, not run on hardware
//
#include <hip/hip_runtime.h>

typedef __attribute__((ext_vector_type(16))) _Float16 v16h;
typedef __attribute__((ext_vector_type(8)))  float    v8f;

#define NROWS 16384
#define NCLS  1024
#define DDIM  4096
#define KTILES (DDIM / 32)   // 128

// ---------------------------------------------------------------------------
// Kernel 1: per-row inverse L2 norms for both matrices (eps = 1e-8)
// ---------------------------------------------------------------------------
__global__ void rownorm_kernel(const float* __restrict__ hvs,
                               const float* __restrict__ am,
                               float* __restrict__ invA,
                               float* __restrict__ invB) {
  __shared__ float red[256];
  const int row = blockIdx.x;
  const float* p;
  float* out;
  int orow;
  if (row < NROWS) { p = hvs + (size_t)row * DDIM;           out = invA; orow = row; }
  else             { p = am  + (size_t)(row - NROWS) * DDIM; out = invB; orow = row - NROWS; }
  float s = 0.0f;
  for (int i = threadIdx.x; i < DDIM; i += 256) { float v = p[i]; s += v * v; }
  red[threadIdx.x] = s;
  __syncthreads();
  for (int off = 128; off > 0; off >>= 1) {
    if ((int)threadIdx.x < off) red[threadIdx.x] += red[threadIdx.x + off];
    __syncthreads();
  }
  if (threadIdx.x == 0) out[orow] = 1.0f / fmaxf(sqrtf(red[0]), 1e-8f);
}

// Pack 16 fp32 values into a v16h fragment (element j = j-th value).
__device__ __forceinline__ v16h pack16(float4 a, float4 b, float4 c, float4 d) {
  v16h r;
  r[0]  = (_Float16)a.x; r[1]  = (_Float16)a.y; r[2]  = (_Float16)a.z; r[3]  = (_Float16)a.w;
  r[4]  = (_Float16)b.x; r[5]  = (_Float16)b.y; r[6]  = (_Float16)b.z; r[7]  = (_Float16)b.w;
  r[8]  = (_Float16)c.x; r[9]  = (_Float16)c.y; r[10] = (_Float16)c.z; r[11] = (_Float16)c.w;
  r[12] = (_Float16)d.x; r[13] = (_Float16)d.y; r[14] = (_Float16)d.z; r[15] = (_Float16)d.w;
  return r;
}

// ---------------------------------------------------------------------------
// Kernel 2: convert am to f16 ONCE, pre-swizzled into B-fragment order:
//   amh[((cTile*KTILES + kTile)*32 + lane)*16 + j]
//     = am[cTile*16 + (lane&15)][kTile*32 + (lane>>4)*16 + j]
// so each GEMM wave reads one contiguous 1KB chunk per (tile, k-step).
// ---------------------------------------------------------------------------
__global__ void am_swizzle_kernel(const float* __restrict__ am,
                                  _Float16* __restrict__ amh) {
  const int idx   = blockIdx.x * 256 + threadIdx.x;   // 64*128*32 entries
  const int lane  = idx & 31;
  const int kTile = (idx >> 5) & (KTILES - 1);
  const int cTile = idx >> 12;
  const int col = cTile * 16 + (lane & 15);
  const int k0  = kTile * 32 + (lane >> 4) * 16;
  const float4* src = (const float4*)(am + (size_t)col * DDIM + k0);
  const v16h h = pack16(src[0], src[1], src[2], src[3]);
  *(v16h*)(amh + (size_t)idx * 16) = h;
}

// ---------------------------------------------------------------------------
// Kernel 3: WMMA GEMM. One block (8 wave32) per 16-row stripe covers all 1024
// columns; hvs stripe converted to f16 in LDS (A-fragment order) once. K loop
// is software-pipelined (double-buffered A/B register fragments) so the 18
// loads for step k+1 are in flight while the 8 WMMAs of step k execute.
// Fused argmax/eta epilogue.
// ---------------------------------------------------------------------------
__global__ __launch_bounds__(256)
void cosine_argmax_kernel(const float* __restrict__ hvs,
                          const _Float16* __restrict__ amh,
                          const float* __restrict__ invA,
                          const float* __restrict__ invB,
                          int* __restrict__ preds,
                          float* __restrict__ eta) {
  // A stripe in fragment order: sA[(kTile*32 + lane)*16 + j]
  //   = f16(hvs[m0 + (lane&15)][kTile*32 + (lane>>4)*16 + j])
  __shared__ _Float16 sA[KTILES * 32 * 16];           // 128 KB
  __shared__ float sVal[8][16];
  __shared__ int   sIdx[8][16];
  __shared__ float s01[16][2];

  const int tid  = threadIdx.x;
  const int wave = tid >> 5;
  const int lane = tid & 31;
  const int l16  = lane & 15;
  const int half = lane >> 4;

  const int m0      = blockIdx.x << 4;   // 16 rows per block
  const int colBase = wave << 7;         // 128 cols per wave

  // ---- stage hvs stripe into LDS as f16 A-fragments (once per block) ----
  for (int e = tid; e < KTILES * 32; e += 256) {
    const int eLane = e & 31;
    const int eKt   = e >> 5;
    const int row   = eLane & 15;
    const int gk    = eKt * 32 + (eLane >> 4) * 16;
    const float4* p0 = (const float4*)(hvs + (size_t)(m0 + row) * DDIM + gk);
    const float4* p1 = (const float4*)(hvs + (size_t)(m0 + row) * DDIM + gk + 16);
    const v16h h = pack16(p0[0], p0[1], p0[2], p0[3]);
    *(v16h*)(sA + (size_t)e * 16) = h;
  }
  __syncthreads();

  v8f acc[8] = {};

  // B base pointers: tile (wave*8+t) starts at amh + tileIdx*KTILES*512
  const _Float16* bp[8];
#pragma unroll
  for (int t = 0; t < 8; ++t)
    bp[t] = amh + (size_t)((wave * 8 + t) * KTILES) * 512 + lane * 16;

  const _Float16* ap = sA + lane * 16;

  // ---- software-pipelined main loop (2x unroll, ping-pong buffers) ----
  v16h af0, af1, bf0[8], bf1[8];
  af0 = *(const v16h*)(ap);
#pragma unroll
  for (int t = 0; t < 8; ++t) bf0[t] = *(const v16h*)(bp[t]);

#pragma unroll 1
  for (int kt = 0; kt < KTILES; kt += 2) {
    // prefetch step kt+1 into buffer 1
    af1 = *(const v16h*)(ap + (kt + 1) * 512);
#pragma unroll
    for (int t = 0; t < 8; ++t) bf1[t] = *(const v16h*)(bp[t] + (kt + 1) * 512);

    // compute step kt from buffer 0
#pragma unroll
    for (int t = 0; t < 8; ++t)
      acc[t] = __builtin_amdgcn_wmma_f32_16x16x32_f16(
          false, af0, false, bf0[t], (short)0, acc[t], false, false);

    // prefetch step kt+2 into buffer 0 (wrap on last iter: harmless reload)
    const int kn = (kt + 2) & (KTILES - 1);
    af0 = *(const v16h*)(ap + kn * 512);
#pragma unroll
    for (int t = 0; t < 8; ++t) bf0[t] = *(const v16h*)(bp[t] + kn * 512);

    // compute step kt+1 from buffer 1
#pragma unroll
    for (int t = 0; t < 8; ++t)
      acc[t] = __builtin_amdgcn_wmma_f32_16x16x32_f16(
          false, af1, false, bf1[t], (short)0, acc[t], false, false);
  }

  // ---- epilogue: scale by invA[m]*invB[n], argmax per row, eta ----
  // C/D layout: acc[t][r] -> M = r + half*8, N = colBase + t*16 + l16
  float ia[8];
#pragma unroll
  for (int r = 0; r < 8; ++r) ia[r] = invA[m0 + half * 8 + r];
  float ib[8];
#pragma unroll
  for (int t = 0; t < 8; ++t) ib[t] = invB[colBase + t * 16 + l16];

  // sims[:,0] and sims[:,1] live in wave 0, tile 0, lanes {0,16} / {1,17}
  if (wave == 0 && l16 < 2) {
#pragma unroll
    for (int r = 0; r < 8; ++r)
      s01[half * 8 + r][l16] = acc[0][r] * ia[r] * ib[0];
  }

#pragma unroll
  for (int r = 0; r < 8; ++r) {
    float best = -3.402823466e38f;
    int   bidx = 0;
#pragma unroll
    for (int t = 0; t < 8; ++t) {
      const float v   = acc[t][r] * ia[r] * ib[t];
      const int   col = colBase + t * 16 + l16;
      if (v > best || (v == best && col < bidx)) { best = v; bidx = col; }
    }
    // butterfly reduction within each 16-lane half (xor masks stay in-half)
#pragma unroll
    for (int off = 8; off >= 1; off >>= 1) {
      const float ov = __shfl_xor(best, off, 32);
      const int   oi = __shfl_xor(bidx, off, 32);
      if (ov > best || (ov == best && oi < bidx)) { best = ov; bidx = oi; }
    }
    if (l16 == 0) {
      sVal[wave][half * 8 + r] = best;
      sIdx[wave][half * 8 + r] = bidx;
    }
  }
  __syncthreads();

  if (tid < 16) {
    float best = sVal[0][tid];
    int   bidx = sIdx[0][tid];
#pragma unroll
    for (int w = 1; w < 8; ++w) {
      const float v = sVal[w][tid];
      const int   i = sIdx[w][tid];
      if (v > best || (v == best && i < bidx)) { best = v; bidx = i; }
    }
    preds[m0 + tid] = bidx;                                   // int32 preds
    eta[m0 + tid]   = (s01[tid][1] - s01[tid][0]) * 0.25f + 0.5f;
  }
}

extern "C" void kernel_launch(void* const* d_in, const int* in_sizes, int n_in,
                              void* d_out, int out_size, void* d_ws, size_t ws_size,
                              hipStream_t stream) {
  (void)in_sizes; (void)n_in; (void)out_size; (void)ws_size;
  const float* hvs = (const float*)d_in[0];   // [16384, 4096] f32
  const float* am  = (const float*)d_in[1];   // [1024, 4096]  f32

  float*    invA = (float*)d_ws;                        // [16384]
  float*    invB = invA + NROWS;                        // [1024]
  _Float16* amh  = (_Float16*)((char*)d_ws + 69632);    // 8 MB, 32B-aligned

  int*   preds = (int*)d_out;                 // first N words: int32 preds
  float* eta   = (float*)d_out + NROWS;       // next  N words: f32 eta

  rownorm_kernel<<<NROWS + NCLS, 256, 0, stream>>>(hvs, am, invA, invB);
  am_swizzle_kernel<<<(NCLS / 16) * KTILES * 32 / 256, 256, 0, stream>>>(am, amh);
  cosine_argmax_kernel<<<NROWS / 16, 256, 0, stream>>>(hvs, amh, invA, invB, preds, eta);
}
